// MultiHeadAttention_54331336294677
// MI455X (gfx1250) — matrix operand order repaired and checked by reading, not compiled, not run on hardware
//
#include <hip/hip_runtime.h>

#define BATCH  4
#define SEQ    2048
#define DMODEL 1024
#define NHEADS 16
#define DKH    64
#define ROWS   (BATCH * SEQ)

typedef __bf16 bf16;
typedef bf16  v16bf __attribute__((ext_vector_type(16)));
typedef bf16  v8bf  __attribute__((ext_vector_type(8)));
typedef float v8f   __attribute__((ext_vector_type(8)));
typedef int   i32x4 __attribute__((ext_vector_type(4)));
typedef int   i32x8 __attribute__((ext_vector_type(8)));

__device__ __forceinline__ v8f v8f_zero() {
  v8f z = {0.f, 0.f, 0.f, 0.f, 0.f, 0.f, 0.f, 0.f};
  return z;
}

__device__ __forceinline__ v8f wmma_bf16(v16bf a, v16bf b, v8f c) {
  return __builtin_amdgcn_wmma_f32_16x16x32_bf16(
      false, a, false, b, (short)0, c, false, false);
}

// A-matrix 16x32 bf16 per-lane fragment (ISA 7.12.2).
__device__ __forceinline__ v16bf make_a_frag(const bf16* rowp, int kofs) {
  v8bf lo = *(const v8bf*)(rowp + kofs * 2);
  v8bf hi = *(const v8bf*)(rowp + kofs * 2 + 8);
  v16bf a;
#pragma unroll
  for (int i = 0; i < 8; ++i) { a[i] = lo[i]; a[8 + i] = hi[i]; }
  return a;
}

// ---- CDNA5 async global->LDS copy (ASYNCcnt path) -----------------------
__device__ __forceinline__ void async_copy_b128(void* lds_dst, const void* gsrc) {
  uint32_t l = (uint32_t)(uintptr_t)lds_dst;   // low 32 bits = LDS byte address
  uint64_t g = (uint64_t)(uintptr_t)gsrc;
  asm volatile("global_load_async_to_lds_b128 %0, %1, off"
               :: "v"(l), "v"(g) : "memory");
}
__device__ __forceinline__ void async_wait0() {
  asm volatile("s_wait_asynccnt 0" ::: "memory");
}

// ---- CDNA5 Tensor Data Mover: 2D tile (64 rows x 128B) -> LDS -----------
// Rows are 32 dwords; LDS rows padded by 4 dwords every 32 (-> 144B stride).
__device__ __forceinline__ void tdm_load_tile_2d(uint32_t lds_off,
                                                 const void* gptr,
                                                 uint32_t row_stride_dw) {
  uint64_t ga = (uint64_t)(uintptr_t)gptr;
  i32x4 g0;
  g0[0] = 1;                                              // count=1 (user D#)
  g0[1] = (int)lds_off;                                   // lds_addr
  g0[2] = (int)(uint32_t)ga;                              // global_addr[31:0]
  g0[3] = (int)(((ga >> 32) & 0x01FFFFFFu) | (2u << 30)); // addr[56:32], type=2
  i32x8 g1;
  g1[0] = (2 << 16) | (1 << 20) | (4 << 22) | (3 << 25);  // 4B elems, pad 4dw/32dw
  g1[1] = (32 << 16);                                     // tensor_dim0 = 32 dw
  g1[2] = (64 << 16);                                     // tensor_dim1 = 64 rows
  g1[3] = (32 << 16);                                     // tile_dim0 = 32 dw
  g1[4] = 64;                                             // tile_dim1 = 64 rows
  g1[5] = (int)row_stride_dw;                             // tensor_dim0_stride
  g1[6] = 0;
  g1[7] = 0;
  asm volatile("tensor_load_to_lds %0, %1" :: "s"(g0), "s"(g1) : "memory");
}

// ---------------------------------------------------------------- convert
__global__ __launch_bounds__(256) void cvt_f32_bf16(const float* __restrict__ in,
                                                    bf16* __restrict__ out, int n) {
  for (int i = blockIdx.x * blockDim.x + threadIdx.x; i < n;
       i += gridDim.x * blockDim.x)
    out[i] = (bf16)in[i];
}

// ---------------------------------------------------------------- QKV GEMM
// Out[m,n] = sum_k X[m,k]*W[n,k] + bias[n]  (X fp32, W bf16, Out bf16)
// 256 threads = 8 waves; block tile 128x128; wave tile 32x64 (8 WMMA / k-chunk).
__global__ __launch_bounds__(256) void qkv_gemm_kernel(
    const float* __restrict__ X, const bf16* __restrict__ Wb,
    const float* __restrict__ bias, bf16* __restrict__ Out, int vtransposed) {
  __shared__ bf16 lA[128][40];   // row stride 80B (16B aligned)
  __shared__ bf16 lB[128][40];

  const int m0   = blockIdx.x * 128;
  const int n0   = blockIdx.y * 128;
  const int tid  = threadIdx.x;
  const int wave = tid >> 5;
  const int lane = tid & 31;
  const int hlf  = lane >> 4;
  const int ln16 = lane & 15;
  const int wm   = wave & 3;   // M sub-tile (32 rows)
  const int wn   = wave >> 2;  // N sub-tile (64 cols)

  const int srow = tid >> 1;        // staging: 2 threads per row
  const int scol = (tid & 1) * 16;  // 16 elements each

  v8f acc[2][4];
#pragma unroll
  for (int i = 0; i < 2; ++i)
#pragma unroll
    for (int t = 0; t < 4; ++t) acc[i][t] = v8f_zero();

  for (int kk = 0; kk < DMODEL; kk += 32) {
    __syncthreads();
    // stage A: fp32 -> bf16 (needs conversion, keep VGPR path)
    const float* xp = X + (size_t)(m0 + srow) * DMODEL + kk + scol;
#pragma unroll
    for (int j = 0; j < 4; ++j) {
      float4 f = *(const float4*)(xp + j * 4);
      bf16* dst = &lA[srow][scol + j * 4];
      dst[0] = (bf16)f.x; dst[1] = (bf16)f.y; dst[2] = (bf16)f.z; dst[3] = (bf16)f.w;
    }
    // stage B: bf16 weights via async global->LDS (2 x 16B per thread)
    const bf16* wp = Wb + (size_t)(n0 + srow) * DMODEL + kk + scol;
    async_copy_b128(&lB[srow][scol], wp);
    async_copy_b128(&lB[srow][scol + 8], wp + 8);
    if (kk + 32 < DMODEL) {
      __builtin_prefetch(xp + 32, 0, 1);
      __builtin_prefetch(wp + 32, 0, 1);
    }
    async_wait0();
    __syncthreads();

    v16bf a0 = make_a_frag(&lA[wm * 32 + ln16][0], hlf * 8);
    v16bf a1 = make_a_frag(&lA[wm * 32 + 16 + ln16][0], hlf * 8);
#pragma unroll
    for (int t = 0; t < 4; ++t) {
      v16bf bfr = *(const v16bf*)&lB[wn * 64 + t * 16 + ln16][hlf * 16];
      acc[0][t] = wmma_bf16(a0, bfr, acc[0][t]);
      acc[1][t] = wmma_bf16(a1, bfr, acc[1][t]);
    }
  }

#pragma unroll
  for (int i = 0; i < 2; ++i)
#pragma unroll
    for (int t = 0; t < 4; ++t) {
      const int ncol = n0 + wn * 64 + t * 16 + ln16;
      const float bv = bias[ncol];
      const int h = ncol >> 6, d = ncol & 63;
#pragma unroll
      for (int rr = 0; rr < 8; ++rr) {
        const int mrow = m0 + wm * 32 + i * 16 + rr + hlf * 8;
        const int b = mrow >> 11, s = mrow & (SEQ - 1);
        const float val = acc[i][t][rr] + bv;
        const size_t idx = vtransposed
            ? ((((size_t)b * NHEADS + h) * DKH + d) * SEQ + s)
            : ((((size_t)b * NHEADS + h) * SEQ + s) * DKH + d);
        Out[idx] = (bf16)val;
      }
    }
}

// ---------------------------------------------------------------- attention
// One block = one (b,h), 64 query rows; 4 waves x 16 rows; flash softmax.
// K/V tiles fetched by the Tensor Data Mover; Q staged with async copies.
__global__ __launch_bounds__(128) void attn_kernel(
    const bf16* __restrict__ Q, const bf16* __restrict__ K,
    const bf16* __restrict__ VT, bf16* __restrict__ Ctx) {
  __shared__ bf16 lQ[64][72];       // [qrow][d]  stride 144B
  __shared__ bf16 lK[64][72];       // [key][d]   (TDM-filled, padded rows)
  __shared__ bf16 lV[64][72];       // [d][key]   (V transposed, TDM-filled)
  __shared__ bf16 lP[4][16][72];    // per-wave P re-layout patch

  const int bh  = blockIdx.y;
  const int b   = bh >> 4, h = bh & 15;
  const int q0  = blockIdx.x * 64;
  const int tid = threadIdx.x;
  const int wave = tid >> 5, lane = tid & 31;
  const int hlf = lane >> 4, ln16 = lane & 15;

  const int srow = tid >> 1;
  const int scol = (tid & 1) * 32;

  const bf16* qbase = Q  + (size_t)bh * SEQ * DKH;
  const bf16* kbase = K  + (size_t)bh * SEQ * DKH;
  const bf16* vbase = VT + (size_t)bh * DKH * SEQ;

  // stage Q tile once (async copies, 4 x 16B per thread)
  {
    const bf16* src = qbase + (size_t)(q0 + srow) * DKH + scol;
#pragma unroll
    for (int j = 0; j < 4; ++j)
      async_copy_b128(&lQ[srow][scol + j * 8], src + j * 8);
  }
  async_wait0();
  __syncthreads();
  v16bf qA[2];
#pragma unroll
  for (int kc = 0; kc < 2; ++kc)
    qA[kc] = make_a_frag(&lQ[wave * 16 + ln16][kc * 32], hlf * 8);

  float mrow[8], lrow[8];
  v8f oacc[4];
#pragma unroll
  for (int i = 0; i < 8; ++i) { mrow[i] = -1e30f; lrow[i] = 0.0f; }
#pragma unroll
  for (int t = 0; t < 4; ++t) oacc[t] = v8f_zero();

  const uint32_t lK_off = (uint32_t)(uintptr_t)&lK[0][0];
  const uint32_t lV_off = (uint32_t)(uintptr_t)&lV[0][0];

  for (int kb = 0; kb < SEQ; kb += 64) {
    __syncthreads();  // everyone done with previous lK/lV
    if (wave == 0) {
      // K tile: 64 keys x 64 d, contiguous rows (32 dw stride)
      tdm_load_tile_2d(lK_off, kbase + (size_t)kb * DKH, 32);
      // V^T tile: 64 d-rows x 64 keys, row stride = SEQ bf16 = 1024 dw
      tdm_load_tile_2d(lV_off, vbase + kb, SEQ / 2);
      __builtin_amdgcn_s_wait_tensorcnt(0);
    }
    __syncthreads();

    // S = Q x K^T
    v8f sc[4];
#pragma unroll
    for (int t = 0; t < 4; ++t) sc[t] = v8f_zero();
#pragma unroll
    for (int kc = 0; kc < 2; ++kc) {
#pragma unroll
      for (int t = 0; t < 4; ++t) {
        v16bf bfr = *(const v16bf*)&lK[t * 16 + ln16][kc * 32 + hlf * 16];
        sc[t] = wmma_bf16(qA[kc], bfr, sc[t]);
      }
    }
#pragma unroll
    for (int t = 0; t < 4; ++t) sc[t] *= 0.125f;  // 1/sqrt(64)

    // online softmax (row's 16 values live in one 16-lane half)
#pragma unroll
    for (int rr = 0; rr < 8; ++rr) {
      float mx = fmaxf(fmaxf(sc[0][rr], sc[1][rr]), fmaxf(sc[2][rr], sc[3][rr]));
#pragma unroll
      for (int mask = 1; mask < 16; mask <<= 1)
        mx = fmaxf(mx, __shfl_xor(mx, mask, 32));
      const float mnew = fmaxf(mrow[rr], mx);
      const float corr = __expf(mrow[rr] - mnew);
      mrow[rr] = mnew;
      float ps = 0.0f;
#pragma unroll
      for (int t = 0; t < 4; ++t) {
        const float p = __expf(sc[t][rr] - mnew);
        sc[t][rr] = p;
        ps += p;
      }
#pragma unroll
      for (int mask = 1; mask < 16; mask <<= 1)
        ps += __shfl_xor(ps, mask, 32);
      lrow[rr] = lrow[rr] * corr + ps;
#pragma unroll
      for (int t = 0; t < 4; ++t) oacc[t][rr] *= corr;
    }

    // re-layout P (C-layout -> A-layout) through per-wave LDS patch
#pragma unroll
    for (int t = 0; t < 4; ++t)
#pragma unroll
      for (int rr = 0; rr < 8; ++rr)
        lP[wave][rr + hlf * 8][t * 16 + ln16] = (bf16)sc[t][rr];
    __syncthreads();

    // O += P x V
#pragma unroll
    for (int kc = 0; kc < 2; ++kc) {
      v16bf pA = make_a_frag(&lP[wave][ln16][kc * 32], hlf * 8);
#pragma unroll
      for (int t = 0; t < 4; ++t) {
        v16bf bfr = *(const v16bf*)&lV[t * 16 + ln16][kc * 32 + hlf * 16];
        oacc[t] = wmma_bf16(pA, bfr, oacc[t]);
      }
    }
  }

  // normalize + merge heads -> [B,S,DMODEL] bf16
#pragma unroll
  for (int t = 0; t < 4; ++t) {
    const int d = t * 16 + ln16;
#pragma unroll
    for (int rr = 0; rr < 8; ++rr) {
      const int srowq = q0 + wave * 16 + rr + hlf * 8;
      const float val = oacc[t][rr] / lrow[rr];
      Ctx[((size_t)b * SEQ + srowq) * DMODEL + h * DKH + d] = (bf16)val;
    }
  }
}

// ---------------------------------------------------------------- output GEMM
__global__ __launch_bounds__(256) void out_gemm_kernel(
    const bf16* __restrict__ Xb, const bf16* __restrict__ Wb,
    const float* __restrict__ bias, float* __restrict__ Out) {
  __shared__ bf16 lA[128][40];
  __shared__ bf16 lB[128][40];

  const int m0   = blockIdx.x * 128;
  const int n0   = blockIdx.y * 128;
  const int tid  = threadIdx.x;
  const int wave = tid >> 5;
  const int lane = tid & 31;
  const int hlf  = lane >> 4;
  const int ln16 = lane & 15;
  const int wm   = wave & 3;
  const int wn   = wave >> 2;

  const int srow = tid >> 1;
  const int scol = (tid & 1) * 16;

  v8f acc[2][4];
#pragma unroll
  for (int i = 0; i < 2; ++i)
#pragma unroll
    for (int t = 0; t < 4; ++t) acc[i][t] = v8f_zero();

  for (int kk = 0; kk < DMODEL; kk += 32) {
    __syncthreads();
    const bf16* xp = Xb + (size_t)(m0 + srow) * DMODEL + kk + scol;
    async_copy_b128(&lA[srow][scol], xp);
    async_copy_b128(&lA[srow][scol + 8], xp + 8);
    const bf16* wp = Wb + (size_t)(n0 + srow) * DMODEL + kk + scol;
    async_copy_b128(&lB[srow][scol], wp);
    async_copy_b128(&lB[srow][scol + 8], wp + 8);
    if (kk + 32 < DMODEL) {
      __builtin_prefetch(xp + 32, 0, 1);
      __builtin_prefetch(wp + 32, 0, 1);
    }
    async_wait0();
    __syncthreads();

    v16bf a0 = make_a_frag(&lA[wm * 32 + ln16][0], hlf * 8);
    v16bf a1 = make_a_frag(&lA[wm * 32 + 16 + ln16][0], hlf * 8);
#pragma unroll
    for (int t = 0; t < 4; ++t) {
      v16bf bfr = *(const v16bf*)&lB[wn * 64 + t * 16 + ln16][hlf * 16];
      acc[0][t] = wmma_bf16(a0, bfr, acc[0][t]);
      acc[1][t] = wmma_bf16(a1, bfr, acc[1][t]);
    }
  }

#pragma unroll
  for (int i = 0; i < 2; ++i)
#pragma unroll
    for (int t = 0; t < 4; ++t) {
      const int ncol = n0 + wn * 64 + t * 16 + ln16;
      const float bv = bias[ncol];
#pragma unroll
      for (int rr = 0; rr < 8; ++rr) {
        const int mrow = m0 + wm * 32 + i * 16 + rr + hlf * 8;
        Out[(size_t)mrow * DMODEL + ncol] = acc[i][t][rr] + bv;
      }
    }
}

// ---------------------------------------------------------------- launch
extern "C" void kernel_launch(void* const* d_in, const int* in_sizes, int n_in,
                              void* d_out, int out_size, void* d_ws, size_t ws_size,
                              hipStream_t stream) {
  const float* q  = (const float*)d_in[0];
  const float* k  = (const float*)d_in[1];
  const float* v  = (const float*)d_in[2];
  const float* Wq = (const float*)d_in[3];
  const float* bq = (const float*)d_in[4];
  const float* Wk = (const float*)d_in[5];
  const float* bk = (const float*)d_in[6];
  const float* Wv = (const float*)d_in[7];
  const float* bv = (const float*)d_in[8];
  const float* Wo = (const float*)d_in[9];
  const float* bo = (const float*)d_in[10];

  bf16* ws = (bf16*)d_ws;
  const size_t WSZ = (size_t)DMODEL * DMODEL;
  const size_t HSZ = (size_t)ROWS * DMODEL;
  bf16* wqb  = ws;
  bf16* wkb  = wqb + WSZ;
  bf16* wvb  = wkb + WSZ;
  bf16* wob  = wvb + WSZ;
  bf16* qbuf = wob + WSZ;        // [B,H,S,64]
  bf16* kbuf = qbuf + HSZ;       // [B,H,S,64]
  bf16* vbuf = kbuf + HSZ;       // [B,H,64,S] (transposed)
  bf16* ctx  = vbuf + HSZ;       // [B,S,DMODEL]

  cvt_f32_bf16<<<512, 256, 0, stream>>>(Wq, wqb, (int)WSZ);
  cvt_f32_bf16<<<512, 256, 0, stream>>>(Wk, wkb, (int)WSZ);
  cvt_f32_bf16<<<512, 256, 0, stream>>>(Wv, wvb, (int)WSZ);
  cvt_f32_bf16<<<512, 256, 0, stream>>>(Wo, wob, (int)WSZ);

  dim3 gg(ROWS / 128, DMODEL / 128);
  qkv_gemm_kernel<<<gg, 256, 0, stream>>>(q, wqb, bq, qbuf, 0);
  qkv_gemm_kernel<<<gg, 256, 0, stream>>>(k, wkb, bk, kbuf, 0);
  qkv_gemm_kernel<<<gg, 256, 0, stream>>>(v, wvb, bv, vbuf, 1);

  dim3 ga(SEQ / 64, BATCH * NHEADS);
  attn_kernel<<<ga, 128, 0, stream>>>(qbuf, kbuf, vbuf, ctx);

  out_gemm_kernel<<<gg, 256, 0, stream>>>(ctx, wob, bo, (float*)d_out);

  (void)in_sizes; (void)n_in; (void)out_size; (void)ws_size;
}
